// CrossModalMoELayer_52046413693427
// MI455X (gfx1250) — compile-verified, matrix operand order checked by reading, not run on hardware
//
#include <hip/hip_runtime.h>
#include <hip/hip_bf16.h>
#include <math.h>

// ---------------------------------------------------------------------------
// CrossModalMoELayer for MI455X (gfx1250, wave32, WMMA 16x16x32 bf16)
// Compute-bound layer (~230 GFLOP vs ~10us of HBM traffic at 23.3 TB/s):
// all GEMMs ride v_wmma_f32_16x16x32_bf16; attention is flash-style so the
// 134MB cross-attn score tensor is never materialized. Tile staging uses
// gfx1250 async global->LDS DMA (ASYNCcnt) when the toolchain exposes it.
// ---------------------------------------------------------------------------

typedef __bf16 bf16_t;
typedef __attribute__((ext_vector_type(16))) __bf16 v16bf;
typedef __attribute__((ext_vector_type(8)))  __bf16 v8bf;
typedef __attribute__((ext_vector_type(8)))  float   v8f;

#define B_   8
#define SQ_  512
#define SI_  1024
#define ST_  256
#define H_   512
#define NH_  8
#define DH_  64
#define E_   8
#define I_   1024

#if defined(__has_builtin)
#if __has_builtin(__builtin_amdgcn_global_load_async_to_lds_b128) && \
    __has_builtin(__builtin_amdgcn_s_wait_asynccnt)
#define HAVE_ASYNC_LDS 1
#endif
#endif

#ifdef HAVE_ASYNC_LDS
// Builtin signature (from hipcc diagnostic): both pointers are to GCC-vector
// int4; src in AS1 (prints as __device__), dst in AS3; then imm offset, imm cpol.
typedef int v4i_gcc __attribute__((vector_size(16)));
typedef __attribute__((address_space(1))) v4i_gcc g1v4i;
typedef __attribute__((address_space(3))) v4i_gcc l3v4i;
// Per-lane 16B global->LDS async DMA (tracked by ASYNCcnt, no VGPR data).
// AS1 pointer value == generic value; AS3 pointer value == low 32 addr bits
// (the flat LDS aperture uses the low 32 address bits).
__device__ __forceinline__ void async_g2l_b128(const void* gsrc, void* ldst) {
  __builtin_amdgcn_global_load_async_to_lds_b128(
      (g1v4i*)(unsigned long long)gsrc,
      (l3v4i*)(unsigned)(unsigned long long)ldst, 0, 0);
}
__device__ __forceinline__ void async_wait0() {
  __builtin_amdgcn_s_wait_asynccnt(0);
}
#endif

__device__ __forceinline__ bf16_t f2bf(float x) {
  union { float f; unsigned u; } cv; cv.f = x;
  unsigned r = (cv.u + 0x7FFFu + ((cv.u >> 16) & 1u)) >> 16;
  union { unsigned short s; bf16_t b; } o; o.s = (unsigned short)r;
  return o.b;
}

__device__ __forceinline__ v8f wmma_bf16(v16bf a, v16bf b, v8f c) {
  return __builtin_amdgcn_wmma_f32_16x16x32_bf16(
      false, a, false, b, (short)0, c, false, false);
}

// ---------------------------------------------------------------------------
// fp32 -> bf16 elementwise conversion (grid-stride)
// ---------------------------------------------------------------------------
__global__ void cvt_bf16_kernel(const float* __restrict__ x,
                                bf16_t* __restrict__ y, size_t n) {
  size_t i = (size_t)blockIdx.x * blockDim.x + threadIdx.x;
  size_t stride = (size_t)gridDim.x * blockDim.x;
  for (; i < n; i += stride) y[i] = f2bf(x[i]);
}

// ---------------------------------------------------------------------------
// LayerNorm over H=512, output bf16 (feeds WMMA GEMMs). One row per block.
// ---------------------------------------------------------------------------
__global__ __launch_bounds__(128)
void layernorm_bf16_kernel(const float* __restrict__ x,
                           const float* __restrict__ g,
                           const float* __restrict__ b,
                           bf16_t* __restrict__ y) {
  const int row = blockIdx.x;
  const int tid = threadIdx.x;
  const float* xr = x + (size_t)row * H_;
  float v[4], s = 0.f, s2 = 0.f;
#pragma unroll
  for (int i = 0; i < 4; ++i) {
    v[i] = xr[tid + i * 128];
    s += v[i]; s2 += v[i] * v[i];
  }
  for (int m = 1; m < 32; m <<= 1) {
    s  += __shfl_xor(s,  m, 32);
    s2 += __shfl_xor(s2, m, 32);
  }
  __shared__ float rs[4], rs2[4];
  const int wave = tid >> 5, lane = tid & 31;
  if (lane == 0) { rs[wave] = s; rs2[wave] = s2; }
  __syncthreads();
  s  = rs[0] + rs[1] + rs[2] + rs[3];
  s2 = rs2[0] + rs2[1] + rs2[2] + rs2[3];
  const float mu  = s / (float)H_;
  const float var = s2 / (float)H_ - mu * mu;
  const float inv = rsqrtf(var + 1e-5f);
#pragma unroll
  for (int i = 0; i < 4; ++i) {
    const int c = tid + i * 128;
    y[(size_t)row * H_ + c] = f2bf((v[i] - mu) * inv * g[c] + b[c]);
  }
}

// ---------------------------------------------------------------------------
// WMMA GEMM: C[M,N] = A[M,K] @ W[N,K]^T (+ bias) with epilogue modes.
//   mode 0: Cf = acc + bias (+ residual[fp32])
//   mode 1: Cb = bf16(acc + bias)
//   mode 2: Cb = bf16(gelu(acc + bias))                 (MoE layer-1)
//   mode 3: Cf += rowscale[row*rs_stride] * (acc+bias)  (MoE layer-2 mix)
// Block: 256 thr = 8 wave32 waves (2x4); each wave a 32x64 tile (2x4 of
// 16x16 WMMA tiles) -> 8 WMMAs per 6 fragment loads per K-step.
// Block tile 64x256, K-step 32, LDS staged via async DMA (ASYNCcnt) when
// available, with global_prefetch of the next K slab.
// ---------------------------------------------------------------------------
__global__ __launch_bounds__(256)
void wmma_gemm_bf16(const bf16_t* __restrict__ A, int lda,
                    const bf16_t* __restrict__ W, int ldw,
                    const float* __restrict__ bias,
                    const float* __restrict__ residual,
                    const float* __restrict__ rowscale, int rs_stride,
                    float* __restrict__ Cf, bf16_t* __restrict__ Cb, int ldc,
                    int M, int N, int K, int mode) {
  __shared__ alignas(32) bf16_t As[64 * 32];    // 4 KB
  __shared__ alignas(32) bf16_t Bs[256 * 32];   // 16 KB
  const int tid  = threadIdx.x;
  const int wave = tid >> 5;
  const int lane = tid & 31;
  const int wm = wave & 1;       // wave M index (0..1)
  const int wn = wave >> 1;      // wave N index (0..3)
  const int bm = blockIdx.y * 64;
  const int bn = blockIdx.x * 256;
  const int hl  = lane >> 4;     // K-half within fragment
  const int l16 = lane & 15;

  v8f acc[2][4] = {};

  const int ar = tid >> 2, ac = (tid & 3) * 8;  // A tile: 64 rows x 32 cols
  const int br = tid;                           // B tile: 256 rows x 32 cols

  for (int k0 = 0; k0 < K; k0 += 32) {
    const bf16_t* asrc = A + (size_t)(bm + ar) * lda + k0 + ac;
    const bf16_t* wsrc = W + (size_t)(bn + br) * ldw + k0;
#ifdef HAVE_ASYNC_LDS
    async_g2l_b128(asrc, &As[ar * 32 + ac]);
#pragma unroll
    for (int c = 0; c < 32; c += 8)
      async_g2l_b128(wsrc + c, &Bs[br * 32 + c]);
#else
    *(v8bf*)&As[ar * 32 + ac] = *(const v8bf*)asrc;
#pragma unroll
    for (int c = 0; c < 32; c += 8)
      *(v8bf*)&Bs[br * 32 + c] = *(const v8bf*)(wsrc + c);
#endif
    if (k0 + 32 < K) {
      __builtin_prefetch(asrc + 32, 0, 1);
      __builtin_prefetch(wsrc + 32, 0, 1);
    }
#ifdef HAVE_ASYNC_LDS
    async_wait0();               // own wave's DMAs landed in LDS
#endif
    __syncthreads();             // everyone's DMAs landed

    v16bf afrag[2], wfrag[4];
#pragma unroll
    for (int i = 0; i < 2; ++i)
      afrag[i] = *(const v16bf*)&As[(wm * 32 + i * 16 + l16) * 32 + hl * 16];
#pragma unroll
    for (int j = 0; j < 4; ++j)
      wfrag[j] = *(const v16bf*)&Bs[(wn * 64 + j * 16 + l16) * 32 + hl * 16];
#pragma unroll
    for (int i = 0; i < 2; ++i)
#pragma unroll
      for (int j = 0; j < 4; ++j)
        acc[i][j] = wmma_bf16(afrag[i], wfrag[j], acc[i][j]);
    __syncthreads();
  }

  // Epilogue: C fragment layout -> lane (l16) = column, VGPR r = row (+8 hi)
#pragma unroll
  for (int i = 0; i < 2; ++i) {
#pragma unroll
    for (int j = 0; j < 4; ++j) {
      const int col = bn + wn * 64 + j * 16 + l16;
#pragma unroll
      for (int r = 0; r < 8; ++r) {
        const int row = bm + wm * 32 + i * 16 + r + hl * 8;
        if (row < M && col < N) {
          float v = acc[i][j][r] + (bias ? bias[col] : 0.f);
          const size_t idx = (size_t)row * ldc + col;
          if (mode == 0) {
            if (residual) v += residual[idx];
            Cf[idx] = v;
          } else if (mode == 1) {
            Cb[idx] = f2bf(v);
          } else if (mode == 2) {
            Cb[idx] = f2bf(0.5f * v * (1.f + erff(v * 0.70710678118654752f)));
          } else {
            Cf[idx] += rowscale[(size_t)row * rs_stride] * v;
          }
        }
      }
    }
  }
}

// ---------------------------------------------------------------------------
// Flash attention: one wave per (batch*head, 16-query tile).
// S = Q Kt via 4 WMMAs / 32-key block; online softmax with shfl_xor
// reductions (layout-coherent with C fragments); P staged via LDS into an
// A fragment; ctx += P V via 4 WMMAs. V tile staged with async DMA issued
// before the QK WMMAs so the copy overlaps matrix ops. Output bf16 ctx.
// ---------------------------------------------------------------------------
__global__ __launch_bounds__(32)
void flash16_kernel(const bf16_t* __restrict__ Qp, int ldq,
                    const bf16_t* __restrict__ Kp, int ldk,
                    const bf16_t* __restrict__ Vp, int ldv,
                    bf16_t* __restrict__ Op, int ldo,
                    int Lq, int Lk, float scale) {
  const int bh = blockIdx.x;
  const int b = bh / NH_, h = bh % NH_;
  const bf16_t* q = Qp + (size_t)b * Lq * ldq + h * DH_;
  const bf16_t* k = Kp + (size_t)b * Lk * ldk + h * DH_;
  const bf16_t* v = Vp + (size_t)b * Lk * ldv + h * DH_;
  bf16_t*       o = Op + (size_t)b * Lq * ldo + h * DH_;
  const int qrow0 = blockIdx.y * 16;
  const int lane = threadIdx.x;
  const int hl = lane >> 4, l16 = lane & 15;

  __shared__ alignas(32) bf16_t Ps[16 * 32];   // P tile (16 q x 32 keys)
  __shared__ alignas(32) bf16_t Vs[32 * 64];   // V tile (32 keys x 64 dh)

  // Q A-fragments: two K=32 slabs covering dh 0..63
  v16bf qa0, qa1;
  {
    const bf16_t* qr = q + (size_t)(qrow0 + l16) * ldq + hl * 16;
    qa0 = *(const v16bf*)qr;
    qa1 = *(const v16bf*)(qr + 32);
  }

  v8f cta[4] = {};
  float mrow[8], lrow[8];
#pragma unroll
  for (int r = 0; r < 8; ++r) { mrow[r] = -1e30f; lrow[r] = 0.f; }

  for (int kb = 0; kb < Lk; kb += 32) {
    // stage V tile (each lane one key row, 64 bf16) -- issue first so the
    // DMA overlaps the QK^T WMMAs below
    const bf16_t* vrow = v + (size_t)(kb + lane) * ldv;
#ifdef HAVE_ASYNC_LDS
#pragma unroll
    for (int c = 0; c < 64; c += 8)
      async_g2l_b128(vrow + c, &Vs[lane * 64 + c]);
#else
#pragma unroll
    for (int c = 0; c < 64; c += 8)
      *(v8bf*)&Vs[lane * 64 + c] = *(const v8bf*)(vrow + c);
#endif

    // K fragments (B operand columns are K-matrix rows: contiguous loads)
    const bf16_t* kr0 = k + (size_t)(kb + l16) * ldk + hl * 16;
    const bf16_t* kr1 = k + (size_t)(kb + 16 + l16) * ldk + hl * 16;
    v16bf k00 = *(const v16bf*)kr0;
    v16bf k01 = *(const v16bf*)(kr0 + 32);
    v16bf k10 = *(const v16bf*)kr1;
    v16bf k11 = *(const v16bf*)(kr1 + 32);
    v8f s0 = {}, s1 = {};
    s0 = wmma_bf16(qa0, k00, s0);
    s0 = wmma_bf16(qa1, k01, s0);
    s1 = wmma_bf16(qa0, k10, s1);
    s1 = wmma_bf16(qa1, k11, s1);

    // online softmax (row stats via 16-lane shfl_xor; matches frag layout)
    float mcur[8], psum[8];
#pragma unroll
    for (int r = 0; r < 8; ++r) {
      float a = fmaxf(s0[r] * scale, s1[r] * scale);
      for (int m = 1; m < 16; m <<= 1) a = fmaxf(a, __shfl_xor(a, m, 32));
      mcur[r] = fmaxf(mrow[r], a);
    }
#pragma unroll
    for (int r = 0; r < 8; ++r) {
      float p0 = __expf(s0[r] * scale - mcur[r]);
      float p1 = __expf(s1[r] * scale - mcur[r]);
      float t = p0 + p1;
      for (int m = 1; m < 16; m <<= 1) t += __shfl_xor(t, m, 32);
      psum[r] = t;
      Ps[(r + hl * 8) * 32 + l16]      = f2bf(p0);
      Ps[(r + hl * 8) * 32 + 16 + l16] = f2bf(p1);
    }
#pragma unroll
    for (int r = 0; r < 8; ++r) {
      const float alpha = __expf(mrow[r] - mcur[r]);
      lrow[r] = lrow[r] * alpha + psum[r];
      mrow[r] = mcur[r];
      cta[0][r] *= alpha; cta[1][r] *= alpha;
      cta[2][r] *= alpha; cta[3][r] *= alpha;
    }
#ifdef HAVE_ASYNC_LDS
    async_wait0();               // V tile DMA complete
#endif
    __syncthreads();  // single-wave WG: lowers to s_nop; LDS in-order per wave

    // P as A fragment (16 x 32 keys)
    v16bf pa = *(const v16bf*)&Ps[l16 * 32 + hl * 16];
    // V as B fragments, one per 16-wide dh tile (gather columns from LDS)
#pragma unroll
    for (int t = 0; t < 4; ++t) {
      v16bf vf;
#pragma unroll
      for (int kk = 0; kk < 16; ++kk)
        vf[kk] = Vs[(hl * 16 + kk) * 64 + t * 16 + l16];
      cta[t] = wmma_bf16(pa, vf, cta[t]);
    }
    __syncthreads();
  }

#pragma unroll
  for (int t = 0; t < 4; ++t)
#pragma unroll
    for (int r = 0; r < 8; ++r) {
      const int row = qrow0 + r + hl * 8;
      o[(size_t)row * ldo + t * 16 + l16] = f2bf(cta[t][r] / lrow[r]);
    }
}

// ---------------------------------------------------------------------------
// Per-batch mean over sequence: x[B,L,H] -> y[B,H]
// ---------------------------------------------------------------------------
__global__ __launch_bounds__(512)
void mean_rows_kernel(const float* __restrict__ x, float* __restrict__ y,
                      int L) {
  const int b = blockIdx.x, hcol = threadIdx.x;
  const float* xb = x + (size_t)b * L * H_;
  float s = 0.f;
  for (int l = 0; l < L; ++l) s += xb[(size_t)l * H_ + hcol];
  y[(size_t)b * H_ + hcol] = s / (float)L;
}

// ---------------------------------------------------------------------------
// Gating: logits[t,e] = x[t]·w[e,:H] + ctx[b]·w[e,H:] + bias[e]; softmax(E=8).
// One wave per token; 8 tokens per 256-thread block.
// ---------------------------------------------------------------------------
__global__ __launch_bounds__(256)
void gate_softmax_kernel(const float* __restrict__ x,
                         const float* __restrict__ ctxv,
                         const float* __restrict__ w,
                         const float* __restrict__ bias,
                         float* __restrict__ probs, int L) {
  const int tok  = blockIdx.x * 8 + (threadIdx.x >> 5);
  const int lane = threadIdx.x & 31;
  const int b = tok / L;
  const float* xr = x + (size_t)tok * H_;
  const float* cr = ctxv + (size_t)b * H_;
  float logit[E_];
#pragma unroll
  for (int e = 0; e < E_; ++e) {
    const float* we = w + (size_t)e * (2 * H_);
    float s = 0.f;
    for (int i = lane; i < H_; i += 32) s += xr[i] * we[i] + cr[i] * we[H_ + i];
    for (int m = 1; m < 32; m <<= 1) s += __shfl_xor(s, m, 32);
    logit[e] = s + bias[e];
  }
  float mx = logit[0];
#pragma unroll
  for (int e = 1; e < E_; ++e) mx = fmaxf(mx, logit[e]);
  float sum = 0.f;
#pragma unroll
  for (int e = 0; e < E_; ++e) { logit[e] = __expf(logit[e] - mx); sum += logit[e]; }
  const float inv = 1.f / sum;
  if (lane == 0)
#pragma unroll
    for (int e = 0; e < E_; ++e) probs[(size_t)tok * E_ + e] = logit[e] * inv;
}

// ---------------------------------------------------------------------------
// Host-side orchestration
// ---------------------------------------------------------------------------
extern "C" void kernel_launch(void* const* d_in, const int* in_sizes, int n_in,
                              void* d_out, int out_size, void* d_ws,
                              size_t ws_size, hipStream_t stream) {
  (void)in_sizes; (void)n_in; (void)out_size; (void)ws_size;

  const float* q_tok    = (const float*)d_in[0];
  const float* img_tok  = (const float*)d_in[1];
  const float* txt_ctx  = (const float*)d_in[2];
  const float* ln_q_g   = (const float*)d_in[3];
  const float* ln_q_b   = (const float*)d_in[4];
  const float* ln_c_g   = (const float*)d_in[5];
  const float* ln_c_b   = (const float*)d_in[6];
  const float* ln_f_g   = (const float*)d_in[7];
  const float* ln_f_b   = (const float*)d_in[8];
  const float* sa_in_w  = (const float*)d_in[9];
  const float* sa_in_b  = (const float*)d_in[10];
  const float* sa_out_w = (const float*)d_in[11];
  const float* sa_out_b = (const float*)d_in[12];
  const float* ca_in_w  = (const float*)d_in[13];
  const float* ca_in_b  = (const float*)d_in[14];
  const float* ca_out_w = (const float*)d_in[15];
  const float* ca_out_b = (const float*)d_in[16];
  const float* ig_w     = (const float*)d_in[17];
  const float* ig_b     = (const float*)d_in[18];
  const float* tg_w     = (const float*)d_in[19];
  const float* tg_b     = (const float*)d_in[20];
  const float* e_w1     = (const float*)d_in[21];
  const float* e_b1     = (const float*)d_in[22];
  const float* e_w2     = (const float*)d_in[23];
  const float* e_b2     = (const float*)d_in[24];

  const int Mq = B_ * SQ_;   // 4096 query tokens
  const int Mi = B_ * SI_;   // 8192 image tokens

  size_t cur = 0;
  auto alloc = [&](size_t bytes) -> void* {
    cur = (cur + 255) & ~(size_t)255;
    void* p = (char*)d_ws + cur;
    cur += bytes;
    return p;
  };

  bf16_t* saw_in  = (bf16_t*)alloc((size_t)3 * H_ * H_ * 2);
  bf16_t* saw_out = (bf16_t*)alloc((size_t)H_ * H_ * 2);
  bf16_t* caw_in  = (bf16_t*)alloc((size_t)3 * H_ * H_ * 2);
  bf16_t* caw_out = (bf16_t*)alloc((size_t)H_ * H_ * 2);
  bf16_t* ew1     = (bf16_t*)alloc((size_t)E_ * I_ * H_ * 2);
  bf16_t* ew2     = (bf16_t*)alloc((size_t)E_ * H_ * I_ * 2);
  bf16_t* img_bf  = (bf16_t*)alloc((size_t)Mi * H_ * 2);
  bf16_t* xnorm   = (bf16_t*)alloc((size_t)Mq * H_ * 2);       // reused 3x
  bf16_t* qkvS    = (bf16_t*)alloc((size_t)Mq * 3 * H_ * 2);
  bf16_t* ctxbf   = (bf16_t*)alloc((size_t)Mq * H_ * 2);       // reused 2x
  bf16_t* qC      = (bf16_t*)alloc((size_t)Mq * H_ * 2);
  bf16_t* kvC     = (bf16_t*)alloc((size_t)Mi * 2 * H_ * 2);
  bf16_t* h1      = (bf16_t*)alloc((size_t)Mi * (size_t)(E_ * I_) * 2); // shared
  float*  mean_i  = (float*)alloc((size_t)B_ * H_ * 4);
  float*  mean_t  = (float*)alloc((size_t)B_ * H_ * 4);
  float*  probs_t = (float*)alloc((size_t)Mq * E_ * 4);
  float*  probs_i = (float*)alloc((size_t)Mi * E_ * 4);

  float* qt      = (float*)d_out;              // [Mq, H]  output 0
  float* img_out = qt + (size_t)Mq * H_;       // [Mi, H]  output 1

  auto cvt = [&](const float* src, bf16_t* dst, size_t n) {
    int blocks = (int)((n + 255) / 256);
    if (blocks > 16384) blocks = 16384;
    cvt_bf16_kernel<<<blocks, 256, 0, stream>>>(src, dst, n);
  };
  auto gemm = [&](const bf16_t* A, int lda, const bf16_t* Wp, int ldw,
                  const float* bias, const float* resid,
                  const float* rsc, int rss,
                  float* Cf, bf16_t* Cb, int ldc,
                  int M, int N, int K, int mode) {
    dim3 g(N / 256, M / 64);
    wmma_gemm_bf16<<<g, 256, 0, stream>>>(A, lda, Wp, ldw, bias, resid,
                                          rsc, rss, Cf, Cb, ldc, M, N, K, mode);
  };

  // ---- weight & activation down-converts ----
  cvt(sa_in_w,  saw_in,  (size_t)3 * H_ * H_);
  cvt(sa_out_w, saw_out, (size_t)H_ * H_);
  cvt(ca_in_w,  caw_in,  (size_t)3 * H_ * H_);
  cvt(ca_out_w, caw_out, (size_t)H_ * H_);
  cvt(e_w1, ew1, (size_t)E_ * I_ * H_);
  cvt(e_w2, ew2, (size_t)E_ * H_ * I_);
  cvt(img_tok, img_bf, (size_t)Mi * H_);

  // ---- self-attention block ----
  layernorm_bf16_kernel<<<Mq, 128, 0, stream>>>(q_tok, ln_q_g, ln_q_b, xnorm);
  gemm(xnorm, H_, saw_in, H_, sa_in_b, nullptr, nullptr, 0,
       nullptr, qkvS, 3 * H_, Mq, 3 * H_, H_, 1);
  flash16_kernel<<<dim3(B_ * NH_, SQ_ / 16), 32, 0, stream>>>(
      qkvS, 3 * H_, qkvS + H_, 3 * H_, qkvS + 2 * H_, 3 * H_,
      ctxbf, H_, SQ_, SQ_, 0.125f);
  gemm(ctxbf, H_, saw_out, H_, sa_out_b, q_tok, nullptr, 0,
       qt, nullptr, H_, Mq, H_, H_, 0);

  // ---- cross-attention block (queries attend to image tokens) ----
  layernorm_bf16_kernel<<<Mq, 128, 0, stream>>>(qt, ln_c_g, ln_c_b, xnorm);
  gemm(xnorm, H_, caw_in, H_, ca_in_b, nullptr, nullptr, 0,
       nullptr, qC, H_, Mq, H_, H_, 1);
  gemm(img_bf, H_, caw_in + (size_t)H_ * H_, H_, ca_in_b + H_, nullptr,
       nullptr, 0, nullptr, kvC, 2 * H_, Mi, 2 * H_, H_, 1);
  flash16_kernel<<<dim3(B_ * NH_, SQ_ / 16), 32, 0, stream>>>(
      qC, H_, kvC, 2 * H_, kvC + H_, 2 * H_, ctxbf, H_, SQ_, SI_, 0.125f);
  gemm(ctxbf, H_, caw_out, H_, ca_out_b, qt, nullptr, 0,
       qt, nullptr, H_, Mq, H_, H_, 0);

  // ---- cross-modal gating ----
  mean_rows_kernel<<<B_, H_, 0, stream>>>(img_tok, mean_i, SI_);
  mean_rows_kernel<<<B_, H_, 0, stream>>>(txt_ctx, mean_t, ST_);
  gate_softmax_kernel<<<Mi / 8, 256, 0, stream>>>(img_tok, mean_t, ig_w, ig_b,
                                                  probs_i, SI_);
  gate_softmax_kernel<<<Mq / 8, 256, 0, stream>>>(qt, mean_i, tg_w, tg_b,
                                                  probs_t, SQ_);

  // ---- MoE, query-token path (accumulates into qt held in d_out) ----
  layernorm_bf16_kernel<<<Mq, 128, 0, stream>>>(qt, ln_f_g, ln_f_b, xnorm);
  gemm(xnorm, H_, ew1, H_, e_b1, nullptr, nullptr, 0,
       nullptr, h1, E_ * I_, Mq, E_ * I_, H_, 2);            // gelu -> bf16
  for (int e = 0; e < E_; ++e)
    gemm(h1 + (size_t)e * I_, E_ * I_, ew2 + (size_t)e * H_ * I_, I_,
         e_b2 + (size_t)e * H_, nullptr, probs_t + e, E_,
         qt, nullptr, H_, Mq, H_, I_, 3);                    // C += p*acc

  // ---- MoE, image-token path ----
  (void)hipMemcpyAsync(img_out, img_tok, (size_t)Mi * H_ * sizeof(float),
                       hipMemcpyDeviceToDevice, stream);
  gemm(img_bf, H_, ew1, H_, e_b1, nullptr, nullptr, 0,
       nullptr, h1, E_ * I_, Mi, E_ * I_, H_, 2);
  for (int e = 0; e < E_; ++e)
    gemm(h1 + (size_t)e * I_, E_ * I_, ew2 + (size_t)e * H_ * I_, I_,
         e_b2 + (size_t)e * H_, nullptr, probs_i + e, E_,
         img_out, nullptr, H_, Mi, H_, I_, 3);
}
